// CapsuleLayer_82248623719109
// MI455X (gfx1250) — matrix operand order repaired and checked by reading, not compiled, not run on hardware
//
#include <hip/hip_runtime.h>
#include <hip/hip_bf16.h>
#include <math.h>
#include <stdint.h>

// ---------------------------------------------------------------------------
// CapsuleLayer dynamic routing, fused-recompute design for MI455X (gfx1250).
//
// Never materializes u_hat (1 GiB). Each routing pass recomputes the needed
// contraction of W via bf16 WMMA (v_wmma_f32_16x16x32_bf16, wave32), making
// the whole problem bound by 5 streaming passes over W (5*256MB ~ 55us at
// 23.3 TB/s) instead of ~6GB of u_hat traffic. W streaming is software-
// pipelined with GLOBAL_LOAD_ASYNC_TO_LDS_B128 + counted s_wait_asynccnt
// (double-buffered LDS), so the WMMA work overlaps the memory stream.
//
// Workspace (all fp32, ~65 MB):
//   c_buf  [J,I,B]  softmax coupling coefficients
//   b_buf  [J,I,B]  routing logits
//   s_buf  [B,J,K]  pre-squash capsule sums
// v is stored in d_out [B,J,K] between iterations (final squash overwrites).
// ---------------------------------------------------------------------------

#define Bc 64
#define Ic 2048
#define Lc 16
#define Jc 64
#define Kc 32

typedef __attribute__((ext_vector_type(16))) __bf16 v16bf;
typedef __attribute__((ext_vector_type(8)))  float  v8f;

// Generic pointers to LDS are {shared-aperture | 32-bit LDS offset}; the
// async-to-LDS VDST operand wants the 32-bit LDS byte address (ISA 10.2).
__device__ __forceinline__ uint32_t lds_off(const void* p)
{
    return (uint32_t)(uintptr_t)p;
}

// GLOBAL_LOAD_ASYNC_TO_LDS_B128 (GV mode): per-lane copy of 16 bytes from a
// 64-bit global address straight into LDS, tracked by ASYNCcnt (no VGPR data).
__device__ __forceinline__ void async_b128(uint32_t lds, const void* g)
{
    asm volatile("global_load_async_to_lds_b128 %0, %1, off"
                 :: "v"(lds), "v"((unsigned long long)(uintptr_t)g)
                 : "memory");
}

template <int N>
__device__ __forceinline__ void wait_async()
{
    asm volatile("s_wait_asynccnt %0" :: "n"(N) : "memory");
}

// ---------------------------------------------------------------------------
// s_pass: s[b,j,k] += sum_i c[b,j,i] * (W[j,i,:,:] @ inp[b,i,:])_k
// One workgroup per (j, i-split). 8 waves: wave -> (b-tile 0..3, k-tile 0..1).
// K-dim of the WMMA = (pair of i) x (L=16) = 32. Double-buffered async stage.
// ---------------------------------------------------------------------------
#define S_SPLIT 8
#define I_PER_WG (Ic / S_SPLIT) // 256

template <bool UNIFORM>
__global__ __launch_bounds__(256) void s_pass_kernel(
    const float* __restrict__ inp,  // [B,I,L]
    const float* __restrict__ W,    // [J,I,K,L]
    const float* __restrict__ cbuf, // [J,I,B] (unused if UNIFORM)
    float* __restrict__ s)          // [B,J,K]
{
    __shared__ float Wlds[2][2 * Kc * Lc]; // [buf][p*512 + k*16 + l]  8 KB
    __shared__ float Ilds[2][Bc * 2 * Lc]; // [buf][b*32 + p*16 + l]  16 KB
    __shared__ float Clds[2][2 * Bc];      // [buf][p*64 + b]          1 KB

    const int j     = blockIdx.x;
    const int split = blockIdx.y;
    const int tid   = threadIdx.x;
    const int lane  = tid & 31;
    const int wave  = tid >> 5;
    const int bt    = wave & 3;  // b-tile (rows of 16)
    const int kt    = wave >> 2; // k-tile (cols of 16)
    const int iBeg  = split * I_PER_WG;

    const int halfsel = lane >> 4; // 0: lanes 0-15, 1: lanes 16-31
    const int nl      = lane & 15; // M for A, N for B

    // async ops issued per wave per stage (uniform across waves by design)
    constexpr int NSTAGE = UNIFORM ? 3 : 4;

    auto issue = [&](int buf, int i0) {
        // W[j, i0:i0+2, :, :] = 1024 contiguous floats: 256 lanes x b128
        async_b128(lds_off(&Wlds[buf][tid * 4]),
                   W + ((size_t)j * Ic + i0) * Kc * Lc + tid * 4);
        // inp[:, i0:i0+2, :] : 32 contiguous floats per b: 2 x b128 per lane
        {
            const int b = tid >> 2, q = tid & 3;
            const float* g = inp + ((size_t)b * Ic + i0) * Lc;
            async_b128(lds_off(&Ilds[buf][b * 32 + q * 4]),      g + q * 4);
            async_b128(lds_off(&Ilds[buf][b * 32 + q * 4 + 16]), g + q * 4 + 16);
        }
        if (!UNIFORM) {
            // 128 floats; every wave writes the same data (keeps ASYNCcnt
            // uniform across waves; identical concurrent LDS writes are benign)
            const int p = lane >> 4, b4 = (lane & 15) * 4;
            async_b128(lds_off(&Clds[buf][p * 64 + b4]),
                       cbuf + ((size_t)j * Ic + (i0 + p)) * Bc + b4);
        }
    };

    v8f acc = {};
    constexpr int NIT = I_PER_WG / 2;

    issue(0, iBeg);
    for (int it = 0; it < NIT; ++it) {
        const int i0  = iBeg + it * 2;
        const int cur = it & 1;
        if (it + 1 < NIT) {
            if (tid == 0) // pull W further ahead toward L2 (global_prefetch_b8)
                __builtin_prefetch(W + ((size_t)j * Ic + i0 + 4) * Kc * Lc, 0, 0);
            issue(cur ^ 1, i0 + 2);      // overlap next stage with compute
            wait_async<NSTAGE>();        // wait for *cur* only (next in flight)
        } else {
            wait_async<0>();
        }
        __syncthreads(); // all waves' cur data visible in LDS

        // Build fragments per ISA 7.12.2 wave32 layouts.
        v16bf afrag, bfrag;
        const int bb = bt * 16 + nl; // A row = batch index
#pragma unroll
        for (int e = 0; e < 16; ++e) {
            // A (16x32 bf16): lanes<16 hold kk 0-7,16-23; lanes>=16 hold 8-15,24-31
            const int kka = halfsel * 8 + (e & 7) + ((e & 8) ? 16 : 0);
            const int pa = kka >> 4, la = kka & 15;
            const float cs = UNIFORM ? (1.0f / (float)Jc) : Clds[cur][pa * 64 + bb];
            afrag[e] = (__bf16)(cs * Ilds[cur][bb * 32 + pa * 16 + la]);
            // B (32x16 bf16): lanes<16 hold kk 0-7,16-23; lanes>=16 hold 8-15,24-31
            const int kkb = halfsel * 8 + (e & 7) + ((e & 8) ? 16 : 0);
            const int pb = kkb >> 4, lb = kkb & 15;
            bfrag[e] = (__bf16)Wlds[cur][pb * 512 + (kt * 16 + nl) * Lc + lb];
        }
        acc = __builtin_amdgcn_wmma_f32_16x16x32_bf16(
            false, afrag, false, bfrag, (short)0, acc, false, false);

        __syncthreads(); // cur may be overwritten by next iteration's issue
    }

    // D layout: VGPR r -> M = r + 8*halfsel ; N = lane&15. Reduce splits w/ atomics.
    const int k = kt * 16 + nl;
#pragma unroll
    for (int r = 0; r < 8; ++r) {
        const int bb = bt * 16 + r + halfsel * 8;
        atomicAdd(&s[((size_t)bb * Jc + j) * Kc + k], acc[r]);
    }
}

// ---------------------------------------------------------------------------
// b_update: blog[j,i,b] (+)= sum_l inp[b,i,l] * T[b,l],
//           T = v[:,j,:] (64x32) x W[j,i,:,:]^T(k,l) (32x16)  via WMMA.
// Wave -> (b-tile 0..3, which-of-2-staged-i 0..1). Double-buffered async stage.
// ---------------------------------------------------------------------------
#define U_SPLIT 16
#define I_PER_WG_U (Ic / U_SPLIT) // 128

template <bool FIRST>
__global__ __launch_bounds__(256) void b_update_kernel(
    const float* __restrict__ inp,  // [B,I,L]
    const float* __restrict__ W,    // [J,I,K,L]
    const float* __restrict__ v,    // [B,J,K]
    float* __restrict__ blog)       // [J,I,B]
{
    __shared__ float Wlds[2][2 * Kc * Lc];
    __shared__ float Ilds[2][Bc * 2 * Lc];

    const int j     = blockIdx.x;
    const int split = blockIdx.y;
    const int tid   = threadIdx.x;
    const int lane  = tid & 31;
    const int wave  = tid >> 5;
    const int bt    = wave & 3;
    const int ip    = wave >> 2; // which of the two staged i's
    const int halfsel = lane >> 4;
    const int nl      = lane & 15;

    constexpr int NSTAGE = 3; // W(1) + inp(2) async ops per wave per stage

    auto issue = [&](int buf, int i0) {
        async_b128(lds_off(&Wlds[buf][tid * 4]),
                   W + ((size_t)j * Ic + i0) * Kc * Lc + tid * 4);
        const int b = tid >> 2, q = tid & 3;
        const float* g = inp + ((size_t)b * Ic + i0) * Lc;
        async_b128(lds_off(&Ilds[buf][b * 32 + q * 4]),      g + q * 4);
        async_b128(lds_off(&Ilds[buf][b * 32 + q * 4 + 16]), g + q * 4 + 16);
    };

    // A = v tile (16 b's x 32 k's), invariant across i: load once.
    v16bf afrag;
    {
        const int bb = bt * 16 + nl;
#pragma unroll
        for (int e = 0; e < 16; ++e) {
            const int kk = halfsel * 8 + (e & 7) + ((e & 8) ? 16 : 0);
            afrag[e] = (__bf16)v[((size_t)bb * Jc + j) * Kc + kk];
        }
    }

    const int iBeg = split * I_PER_WG_U;
    constexpr int NIT = I_PER_WG_U / 2;

    issue(0, iBeg);
    for (int it = 0; it < NIT; ++it) {
        const int i0  = iBeg + it * 2;
        const int cur = it & 1;
        if (it + 1 < NIT) {
            if (tid == 0)
                __builtin_prefetch(W + ((size_t)j * Ic + i0 + 4) * Kc * Lc, 0, 0);
            issue(cur ^ 1, i0 + 2);
            wait_async<NSTAGE>();
        } else {
            wait_async<0>();
        }
        __syncthreads();

        // B = W[j, i0+ip, kk, nl]  (rows kk = k, cols n = l)
        v16bf bfrag;
#pragma unroll
        for (int e = 0; e < 16; ++e) {
            const int kk = halfsel * 8 + (e & 7) + ((e & 8) ? 16 : 0);
            bfrag[e] = (__bf16)Wlds[cur][ip * 512 + kk * Lc + nl];
        }
        v8f t = {};
        t = __builtin_amdgcn_wmma_f32_16x16x32_bf16(
            false, afrag, false, bfrag, (short)0, t, false, false);

        // t: lane nl = l, VGPR r -> b = bt*16 + r + 8*halfsel.
        const int i = i0 + ip;
        float res[8];
#pragma unroll
        for (int r = 0; r < 8; ++r) {
            const int bb = bt * 16 + r + halfsel * 8;
            float prod = t[r] * Ilds[cur][bb * 32 + ip * 16 + nl];
            prod += __shfl_xor(prod, 1, 16);
            prod += __shfl_xor(prod, 2, 16);
            prod += __shfl_xor(prod, 4, 16);
            prod += __shfl_xor(prod, 8, 16);
            res[r] = prod; // agreement[bb, j, i]
        }
        if (nl == 0) {
            const size_t base = ((size_t)j * Ic + i) * Bc;
#pragma unroll
            for (int r = 0; r < 8; ++r) {
                const int bb = bt * 16 + r + halfsel * 8;
                if (FIRST) blog[base + bb] = res[r];
                else       blog[base + bb] += res[r];
            }
        }
        __syncthreads(); // cur may be overwritten by next iteration's issue
    }
}

// ---------------------------------------------------------------------------
// softmax over j (axis 1). Layout [J,I,B]: one thread per (i,b) column, fully
// register-resident (64 values), coalesced across consecutive b.
// ---------------------------------------------------------------------------
__global__ __launch_bounds__(256) void softmax_kernel(
    const float* __restrict__ blog, float* __restrict__ cbuf)
{
    const int idx = blockIdx.x * 256 + threadIdx.x; // over I*B
    const size_t stride = (size_t)Ic * Bc;
    float vals[Jc];
    float m = -INFINITY;
#pragma unroll
    for (int jj = 0; jj < Jc; ++jj) {
        vals[jj] = blog[jj * stride + idx];
        m = fmaxf(m, vals[jj]);
    }
    float sum = 0.f;
#pragma unroll
    for (int jj = 0; jj < Jc; ++jj) {
        vals[jj] = __expf(vals[jj] - m);
        sum += vals[jj];
    }
    const float inv = 1.0f / sum;
#pragma unroll
    for (int jj = 0; jj < Jc; ++jj)
        cbuf[jj * stride + idx] = vals[jj] * inv;
}

// ---------------------------------------------------------------------------
// squash: v = ||s||^2/(1+||s||^2) * s/sqrt(||s||^2+eps), norm over k (=32=wave)
// ---------------------------------------------------------------------------
__global__ __launch_bounds__(256) void squash_kernel(
    const float* __restrict__ s, float* __restrict__ v)
{
    const int row = blockIdx.x * 8 + (threadIdx.x >> 5); // (b*J + j)
    const int k   = threadIdx.x & 31;
    const float x = s[(size_t)row * Kc + k];
    float sq = x * x;
#pragma unroll
    for (int off = 1; off < 32; off <<= 1) sq += __shfl_xor(sq, off, 32);
    const float scale = (sq / (1.0f + sq)) * rsqrtf(sq + 1e-7f);
    v[(size_t)row * Kc + k] = scale * x;
}

__global__ __launch_bounds__(256) void zero_kernel(float* p, int n)
{
    const int i = blockIdx.x * 256 + threadIdx.x;
    if (i < n) p[i] = 0.0f;
}

// ---------------------------------------------------------------------------
extern "C" void kernel_launch(void* const* d_in, const int* in_sizes, int n_in,
                              void* d_out, int out_size, void* d_ws, size_t ws_size,
                              hipStream_t stream)
{
    const float* inp = (const float*)d_in[0]; // [64,2048,16] fp32
    const float* W   = (const float*)d_in[1]; // [64,2048,32,16] fp32
    float* out = (float*)d_out;               // [64,64,32] fp32

    // Workspace carving (~65 MB total).
    float* c_buf = (float*)d_ws;                    // [J,I,B]
    float* b_buf = c_buf + (size_t)Jc * Ic * Bc;    // [J,I,B]
    float* s_buf = b_buf + (size_t)Jc * Ic * Bc;    // [B,J,K]
    const int sN = Bc * Jc * Kc; // 131072

    const dim3 blk(256);
    const dim3 gS(Jc, S_SPLIT);
    const dim3 gU(Jc, U_SPLIT);
    const dim3 gZ((sN + 255) / 256);
    const dim3 gSm((Ic * Bc) / 256);
    const dim3 gSq((Bc * Jc) / 8);

    // ---- routing iteration 0 (c uniform = 1/J, so no softmax needed) ----
    zero_kernel<<<gZ, blk, 0, stream>>>(s_buf, sN);
    s_pass_kernel<true><<<gS, blk, 0, stream>>>(inp, W, nullptr, s_buf);
    squash_kernel<<<gSq, blk, 0, stream>>>(s_buf, out);
    b_update_kernel<true><<<gU, blk, 0, stream>>>(inp, W, out, b_buf);

    // ---- routing iteration 1 ----
    softmax_kernel<<<gSm, blk, 0, stream>>>(b_buf, c_buf);
    zero_kernel<<<gZ, blk, 0, stream>>>(s_buf, sN);
    s_pass_kernel<false><<<gS, blk, 0, stream>>>(inp, W, c_buf, s_buf);
    squash_kernel<<<gSq, blk, 0, stream>>>(s_buf, out);
    b_update_kernel<false><<<gU, blk, 0, stream>>>(inp, W, out, b_buf);

    // ---- routing iteration 2 (final: only s + squash) ----
    softmax_kernel<<<gSm, blk, 0, stream>>>(b_buf, c_buf);
    zero_kernel<<<gZ, blk, 0, stream>>>(s_buf, sN);
    s_pass_kernel<false><<<gS, blk, 0, stream>>>(inp, W, c_buf, s_buf);
    squash_kernel<<<gSq, blk, 0, stream>>>(s_buf, out);
}
